// DifferentialAttention_8555574854224
// MI455X (gfx1250) — matrix operand order repaired and checked by reading, not compiled, not run on hardware
//
#include <hip/hip_runtime.h>
#include <hip/hip_bf16.h>

#define BB   2
#define TT   2048
#define DD   2048
#define HH   16
#define DHD  128

typedef __attribute__((ext_vector_type(16))) __bf16 v16bf;
typedef __attribute__((ext_vector_type(8)))  __bf16 v8bf;
typedef __attribute__((ext_vector_type(8)))  float  v8f;

typedef __attribute__((ext_vector_type(4))) unsigned int u32x4;
typedef __attribute__((ext_vector_type(4))) int          i32x4;
typedef __attribute__((ext_vector_type(8))) int          i32x8;

static __device__ __forceinline__ unsigned short f2bf(float f) {
  union { float f; unsigned u; } x; x.f = f;
  unsigned r = x.u + 0x7FFFu + ((x.u >> 16) & 1u);
  return (unsigned short)(r >> 16);
}

static __device__ __forceinline__ v16bf cat16(v8bf lo, v8bf hi) {
  v16bf o;
#pragma unroll
  for (int i = 0; i < 8; ++i) { o[i] = lo[i]; o[8 + i] = hi[i]; }
  return o;
}

// A fragment (16x32 bf16, M x K), src row-major [row][k], ld in elements.
// ISA layout: lane = row (lane&15); VGPR0-3 hold k = k0+8*half .. +7,
// VGPR4-7 hold k = k0+16+8*half .. +7.  -> two 16B loads per lane.
static __device__ __forceinline__ v16bf load_a(const unsigned short* base, int ld,
                                               int row0, int k0) {
  int lane = threadIdx.x & 31, hf = lane >> 4;
  const __bf16* p = (const __bf16*)base + (size_t)(row0 + (lane & 15)) * ld + k0;
  v8bf lo = *(const v8bf*)(p + hf * 8);
  v8bf hi = *(const v8bf*)(p + 16 + hf * 8);
  return cat16(lo, hi);
}

// B fragment (32x16 bf16, K x N), src stored [n][k] (k contiguous), ld in elements.
// ISA layout: lane = col (lane&15); VGPR0-3 hold k = k0+8*half .. +7,
// VGPR4-7 hold k = k0+16+8*half .. +7.  -> two 16B loads per lane.
static __device__ __forceinline__ v16bf load_b(const unsigned short* base, int ld,
                                               int col0, int k0) {
  int lane = threadIdx.x & 31, hf = lane >> 4;
  const __bf16* p = (const __bf16*)base + (size_t)(col0 + (lane & 15)) * ld + k0;
  v8bf lo = *(const v8bf*)(p + hf * 8);
  v8bf hi = *(const v8bf*)(p + 16 + hf * 8);
  return cat16(lo, hi);
}

#define WMMA_BF16(a, b, c) \
  __builtin_amdgcn_wmma_f32_16x16x32_bf16(false, (a), false, (b), (short)0, (c), false, false)

// ------------------------------------------------------------- TDM tile load
// 2D tile load via Tensor Data Mover: rows x 32 bf16 elements, global row
// stride `ld` elements, packed row-major into LDS at byte offset lds_off.
// D# layout per CDNA5 ISA ch.8 (group0 128b, group1 256b; groups 2/3 zero).
static __device__ __forceinline__ void tdm_load_2d(unsigned lds_off,
                                                   const unsigned short* gptr,
                                                   unsigned rows, unsigned ld) {
  unsigned long long ga = (unsigned long long)(size_t)gptr;
  u32x4 g0;
  g0[0] = 1u;                                            // count=1, user mode
  g0[1] = lds_off;                                       // lds_addr (bytes)
  g0[2] = (unsigned)(ga & 0xFFFFFFFFu);                  // global_addr[31:0]
  g0[3] = (unsigned)((ga >> 32) & 0x01FFFFFFu) | (2u << 30);  // addr[56:32], type=2

  const unsigned td0 = 32u, tile0 = 32u;                 // k extent
  const unsigned td1 = rows, tile1 = rows;
  unsigned long long s0 = (unsigned long long)ld;        // dim0 stride (elements)
  i32x8 g1;
  g1[0] = (int)(1u << 16);                               // data_size=1 (2 bytes)
  g1[1] = (int)((td0 & 0xFFFFu) << 16);                  // tensor_dim0[15:0]
  g1[2] = (int)((td0 >> 16) | ((td1 & 0xFFFFu) << 16));  // td0 hi | td1 lo
  g1[3] = (int)((td1 >> 16) | (tile0 << 16));            // td1 hi | tile_dim0
  g1[4] = (int)(tile1 & 0xFFFFu);                        // tile_dim1 | tile_dim2=0
  g1[5] = (int)(s0 & 0xFFFFFFFFu);                       // dim0_stride[31:0]
  g1[6] = (int)((s0 >> 32) & 0xFFFFu);                   // dim0_stride[47:32]
  g1[7] = 0;

  i32x4 z4 = {0, 0, 0, 0};
#if __clang_major__ >= 23
  i32x8 z8 = {0, 0, 0, 0, 0, 0, 0, 0};
  __builtin_amdgcn_tensor_load_to_lds(g0, g1, z4, z4, z8, 0);
#else
  __builtin_amdgcn_tensor_load_to_lds(g0, g1, z4, z4, 0);
#endif
}

// ---------------------------------------------------------------- cast kernel
__global__ void __launch_bounds__(256) cast_bf16_kernel(const float* __restrict__ in,
                                                        unsigned short* __restrict__ out,
                                                        int n) {
  int i = (blockIdx.x * blockDim.x + threadIdx.x) * 4;
  if (i + 3 < n) {
    float4 v = *(const float4*)(in + i);
    ushort4 o;
    o.x = f2bf(v.x); o.y = f2bf(v.y); o.z = f2bf(v.z); o.w = f2bf(v.w);
    *(ushort4*)(out + i) = o;
  }
}

// ---------------------------------------------------------------- GEMM kernel
// C = A(bf16, MxK row-major) x B(bf16, stored [n][k], NxK row-major)
// Block: 256 thr = 8 waves (4x2), tile 128(M) x 64(N), K-step 32.
// A/B tiles staged in LDS via the Tensor Data Mover, double-buffered.
// mode 0: C = f32 row-major MxN
// mode 1: C = bf16 at [b,h,t,d]   (Q / K layout), m=b*T+t, n=h*128+d
// mode 2: C = bf16 at [b,h,d,t]   (V transposed layout)
__global__ void __launch_bounds__(256) gemm_bf16(const unsigned short* __restrict__ A,
                                                 const unsigned short* __restrict__ Bm,
                                                 void* __restrict__ C,
                                                 int M, int N, int K, int mode) {
  __shared__ unsigned short sA[2][128 * 32];   // 16 KB
  __shared__ unsigned short sB[2][64 * 32];    //  8 KB

  int wave = threadIdx.x >> 5;
  int lane = threadIdx.x & 31, hf = lane >> 4, ln = lane & 15;
  int mb = blockIdx.y * 128;
  int nb = blockIdx.x * 64;
  int wm = (wave >> 1) * 32;     // wave row offset inside block tile
  int wn = (wave & 1) * 32;      // wave col offset inside block tile
  bool w0 = (wave == 0);

  // LDS byte offsets (flat LDS addresses have the offset in the low 32 bits)
  unsigned offA[2] = { (unsigned)(size_t)&sA[0][0], (unsigned)(size_t)&sA[1][0] };
  unsigned offB[2] = { (unsigned)(size_t)&sB[0][0], (unsigned)(size_t)&sB[1][0] };

  v8f acc00 = {}, acc01 = {}, acc10 = {}, acc11 = {};

  const int kSteps = K >> 5;
  if (w0) {
    tdm_load_2d(offA[0], A + (size_t)mb * K, 128, (unsigned)K);
    tdm_load_2d(offB[0], Bm + (size_t)nb * K, 64, (unsigned)K);
  }

  for (int ks = 0; ks < kSteps; ++ks) {
    int cur = ks & 1;
    __syncthreads();                       // everyone done reading buffer cur^1
    if (w0) {
      if (ks + 1 < kSteps) {
        tdm_load_2d(offA[cur ^ 1], A + (size_t)mb * K + (ks + 1) * 32, 128, (unsigned)K);
        tdm_load_2d(offB[cur ^ 1], Bm + (size_t)nb * K + (ks + 1) * 32, 64, (unsigned)K);
        __builtin_amdgcn_s_wait_tensorcnt(2);   // current pair complete (in-order)
      } else {
        __builtin_amdgcn_s_wait_tensorcnt(0);
      }
    }
    __syncthreads();                       // buffer cur visible to all waves

    v16bf a0 = load_a(&sA[cur][0], 32, wm, 0);
    v16bf a1 = load_a(&sA[cur][0], 32, wm + 16, 0);
    v16bf b0 = load_b(&sB[cur][0], 32, wn, 0);
    v16bf b1 = load_b(&sB[cur][0], 32, wn + 16, 0);
    acc00 = WMMA_BF16(a0, b0, acc00);
    acc01 = WMMA_BF16(a0, b1, acc01);
    acc10 = WMMA_BF16(a1, b0, acc10);
    acc11 = WMMA_BF16(a1, b1, acc11);
  }

  v8f accs[2][2] = {{acc00, acc01}, {acc10, acc11}};
#pragma unroll
  for (int i = 0; i < 2; ++i)
#pragma unroll
    for (int j = 0; j < 2; ++j)
#pragma unroll
      for (int v = 0; v < 8; ++v) {
        int m = mb + wm + i * 16 + v + 8 * hf;   // D-layout: comp v <-> row v+8*half
        int n = nb + wn + j * 16 + ln;           //           lane&15 <-> col
        float val = accs[i][j][v];
        if (mode == 0) {
          ((float*)C)[(size_t)m * N + n] = val;
        } else {
          int b = m >> 11, t = m & (TT - 1);
          int h = n >> 7,  d = n & (DHD - 1);
          size_t addr = (mode == 1)
                          ? ((size_t)(b * HH + h) * TT + t) * DHD + d
                          : ((size_t)(b * HH + h) * DHD + d) * TT + t;
          ((unsigned short*)C)[addr] = f2bf(val);
        }
      }
}

// ---------------------------------------------------------- differential attn
// grid = (T/64, H, B), block = 128 (4 waves). Each wave owns 16 query rows.
// Two independent flash-softmax accumulators; combined as O1/l1 - sigmoid(lam)*O2/l2.
__global__ void __launch_bounds__(128) diff_attn_kernel(const unsigned short* __restrict__ Q,
                                                        const unsigned short* __restrict__ Kb,
                                                        const unsigned short* __restrict__ Vt,
                                                        const float* __restrict__ lambda_param,
                                                        unsigned short* __restrict__ out) {
  __shared__ unsigned short lds_p[4 * 2 * 16 * 32];  // per-wave P1/P2 staging (8 KB)

  int w = threadIdx.x >> 5;
  int lane = threadIdx.x & 31, hf = lane >> 4, ln = lane & 15;
  int h = blockIdx.y, b = blockIdx.z;
  int t0 = blockIdx.x * 64 + w * 16;

  const unsigned short* Qh = Q + (size_t)(b * HH + h) * TT * DHD;  // [t][d]
  const unsigned short* Kh = Kb + (size_t)(b * HH + h) * TT * DHD; // [s][d]
  const unsigned short* Vh = Vt + (size_t)(b * HH + h) * DHD * TT; // [d][s]
  unsigned short* P1 = &lds_p[w * 1024];
  unsigned short* P2 = &lds_p[w * 1024 + 512];

  v16bf qa[4];
#pragma unroll
  for (int i = 0; i < 4; ++i) qa[i] = load_a(Qh, DHD, t0, i * 32);

  v8f o1[8] = {}, o2[8] = {};
  float m1[8], l1[8], m2[8], l2[8];
#pragma unroll
  for (int v = 0; v < 8; ++v) { m1[v] = m2[v] = -3.0e38f; l1[v] = l2[v] = 0.f; }

  const float rscale = 0.125f;  // 1/sqrt(D_SUB)

  for (int s0 = 0; s0 < TT; s0 += 32) {
    v8f s1a = {}, s1b = {}, s2a = {}, s2b = {};
#pragma unroll
    for (int kk = 0; kk < 2; ++kk) {
      v16bf bk0 = load_b(Kh, DHD, s0, kk * 32);
      v16bf bk1 = load_b(Kh, DHD, s0 + 16, kk * 32);
      s1a = WMMA_BF16(qa[kk], bk0, s1a);
      s1b = WMMA_BF16(qa[kk], bk1, s1b);
      v16bf ck0 = load_b(Kh, DHD, s0, 64 + kk * 32);
      v16bf ck1 = load_b(Kh, DHD, s0 + 16, 64 + kk * 32);
      s2a = WMMA_BF16(qa[2 + kk], ck0, s2a);
      s2b = WMMA_BF16(qa[2 + kk], ck1, s2b);
    }

    float a1[8], a2[8];
#pragma unroll
    for (int v = 0; v < 8; ++v) {
      int row = v + 8 * hf;
      // ---- branch 1
      {
        float xa = s1a[v] * rscale, xb = s1b[v] * rscale;
        float t = fmaxf(xa, xb);
        for (int off = 1; off < 16; off <<= 1) t = fmaxf(t, __shfl_xor(t, off, 16));
        float mn = fmaxf(m1[v], t);
        float al = __expf(m1[v] - mn);
        float pa = __expf(xa - mn), pb = __expf(xb - mn);
        float rs = pa + pb;
        for (int off = 1; off < 16; off <<= 1) rs += __shfl_xor(rs, off, 16);
        l1[v] = l1[v] * al + rs;
        m1[v] = mn; a1[v] = al;
        P1[row * 32 + ln] = f2bf(pa);
        P1[row * 32 + 16 + ln] = f2bf(pb);
      }
      // ---- branch 2
      {
        float xa = s2a[v] * rscale, xb = s2b[v] * rscale;
        float t = fmaxf(xa, xb);
        for (int off = 1; off < 16; off <<= 1) t = fmaxf(t, __shfl_xor(t, off, 16));
        float mn = fmaxf(m2[v], t);
        float al = __expf(m2[v] - mn);
        float pa = __expf(xa - mn), pb = __expf(xb - mn);
        float rs = pa + pb;
        for (int off = 1; off < 16; off <<= 1) rs += __shfl_xor(rs, off, 16);
        l2[v] = l2[v] * al + rs;
        m2[v] = mn; a2[v] = al;
        P2[row * 32 + ln] = f2bf(pa);
        P2[row * 32 + 16 + ln] = f2bf(pb);
      }
    }

#pragma unroll
    for (int f = 0; f < 8; ++f)
#pragma unroll
      for (int v = 0; v < 8; ++v) { o1[f][v] *= a1[v]; o2[f][v] *= a2[v]; }

    // D-layout -> A-layout via wave-private LDS (ds ops are in-order per wave)
    v16bf pa1 = load_a(P1, 32, 0, 0);
    v16bf pa2 = load_a(P2, 32, 0, 0);

#pragma unroll
    for (int f = 0; f < 8; ++f) {
      v16bf vb = load_b(Vh, TT, f * 16, s0);  // B[n=d][k=s], k contiguous
      o1[f] = WMMA_BF16(pa1, vb, o1[f]);
      o2[f] = WMMA_BF16(pa2, vb, o2[f]);
    }
  }

  float lam = 1.f / (1.f + __expf(-lambda_param[h]));
  float i1[8], i2[8];
#pragma unroll
  for (int v = 0; v < 8; ++v) { i1[v] = 1.f / l1[v]; i2[v] = lam / l2[v]; }

#pragma unroll
  for (int f = 0; f < 8; ++f)
#pragma unroll
    for (int v = 0; v < 8; ++v) {
      int t = t0 + v + 8 * hf;
      int col = h * DHD + f * 16 + ln;
      float val = o1[f][v] * i1[v] - o2[f][v] * i2[v];
      out[(size_t)(b * TT + t) * DD + col] = f2bf(val);
    }
}

// ------------------------------------------------------------------ launcher
extern "C" void kernel_launch(void* const* d_in, const int* in_sizes, int n_in,
                              void* d_out, int out_size, void* d_ws, size_t ws_size,
                              hipStream_t stream) {
  const float* x  = (const float*)d_in[0];
  const float* Wq = (const float*)d_in[1];
  const float* Wk = (const float*)d_in[2];
  const float* Wv = (const float*)d_in[3];
  const float* Wo = (const float*)d_in[4];
  const float* lp = (const float*)d_in[5];

  const size_t NX = (size_t)BB * TT * DD;  // 8,388,608 elements
  const size_t NW = (size_t)DD * DD;       // 4,194,304 elements

  unsigned short* p = (unsigned short*)d_ws;
  unsigned short* xb  = p; p += NX;
  unsigned short* Wqb = p; p += NW;
  unsigned short* Wkb = p; p += NW;
  unsigned short* Wvb = p; p += NW;
  unsigned short* Wob = p; p += NW;
  unsigned short* Qb  = p; p += NX;  // [b,h,t,d]
  unsigned short* Kbq = p; p += NX;  // [b,h,s,d]
  unsigned short* Vtb = p; p += NX;  // [b,h,d,s]
  unsigned short* AOb = p; p += NX;  // [b,t, h*128+d] row-major

  cast_bf16_kernel<<<dim3((unsigned)(NX / 4 / 256)), 256, 0, stream>>>(x, xb, (int)NX);
  cast_bf16_kernel<<<dim3((unsigned)(NW / 4 / 256)), 256, 0, stream>>>(Wq, Wqb, (int)NW);
  cast_bf16_kernel<<<dim3((unsigned)(NW / 4 / 256)), 256, 0, stream>>>(Wk, Wkb, (int)NW);
  cast_bf16_kernel<<<dim3((unsigned)(NW / 4 / 256)), 256, 0, stream>>>(Wv, Wvb, (int)NW);
  cast_bf16_kernel<<<dim3((unsigned)(NW / 4 / 256)), 256, 0, stream>>>(Wo, Wob, (int)NW);

  dim3 gg(DD / 64, (BB * TT) / 128);
  gemm_bf16<<<gg, 256, 0, stream>>>(xb, Wqb, Qb,  BB * TT, DD, DD, 1);
  gemm_bf16<<<gg, 256, 0, stream>>>(xb, Wkb, Kbq, BB * TT, DD, DD, 1);
  gemm_bf16<<<gg, 256, 0, stream>>>(xb, Wvb, Vtb, BB * TT, DD, DD, 2);

  diff_attn_kernel<<<dim3(TT / 64, HH, BB), 128, 0, stream>>>(Qb, Kbq, Vtb, lp, AOb);

  gemm_bf16<<<gg, 256, 0, stream>>>(AOb, Wob, d_out, BB * TT, DD, DD, 0);
}
